// GraphAttentionTemplate_72859825209750
// MI455X (gfx1250) — matrix operand order repaired and checked by reading, not compiled, not run on hardware
//
#include <hip/hip_runtime.h>

// ---------------------------------------------------------------------------
// GraphAttention (additive GAT-style) for MI455X / gfx1250, wave32 + WMMA.
// B=4, N=4096, D=512, H=8, K=64.
//
//   h  = x @ Wa^T                         (WMMA pass A, f16 in / f32 acc)
//   t1 = h.Wb + b1 ; t2 = h.Wc + b2
//   w[n,m] = exp(lrelu(t1[n]+t2[m]) - lrelu(t1[n]+max_m t2[m]))
//   out[n,:] = (sum_m w[n,m] h[m,:]) / (sum_m w[n,m]) + bias
//
// The streaming loops are unrolled x2 with ping-pong operand buffers so the
// double-buffer hand-off is pure register renaming (no v_mov rotation), loads
// are issued a full stage ahead of their consuming WMMAs, and the next weight
// tile's exp/lrelu VALU work covers load latency. A 5th WMMA vs an all-ones
// B matrix produces softmax denominators in the same lane slots as the output
// fragments. global_prefetch warms chunk+2.
// ---------------------------------------------------------------------------

typedef __attribute__((ext_vector_type(16))) _Float16 v16h;
typedef __attribute__((ext_vector_type(8)))  float    v8f;

#define NB 4
#define NN 4096
#define ND 512
#define NK 64

// ---- workspace layout (bytes) ----
#define WS_HT    0          // f16 [B][64][4096]  : 2 MB
#define WS_WA16  2097152    // f16 [64][512]      : 64 KB
#define WS_T1    2162688    // f32 [B][4096]      : 64 KB
#define WS_T2    2228224    // f32 [B][4096]      : 64 KB
#define WS_T2MX  2293760    // f32 [B]

// K-index inside a 16x32 f16 A-fragment for element e (0..15) of a lane.
// Contiguous order: lanes 0-15 carry K 0..15, lanes 16-31 carry K 16..31.
__device__ __forceinline__ int a_frag_k(int e, int hi) {
  return (hi ? 16 : 0) + e;
}

// ---------------------------------------------------------------------------
// Kernel 0: convert Wa (f32 [64][512]) to f16 once.
// ---------------------------------------------------------------------------
__global__ void gat_cvt_wa(const float* __restrict__ Wa, _Float16* __restrict__ wa16) {
  int i = blockIdx.x * 256 + threadIdx.x;          // 32768 elements
  wa16[i] = (_Float16)Wa[i];
}

// ---------------------------------------------------------------------------
// Pass A fragment load: A from x rows (f32->f16), B from pre-converted Wa^T.
// ---------------------------------------------------------------------------
__device__ __forceinline__ void pa_load(const float* __restrict__ xrow,
                                        const _Float16* __restrict__ wrow,
                                        int kb, int hi, int col,
                                        v16h& a, v16h b[4]) {
#pragma unroll
  for (int ct = 0; ct < 4; ++ct)
    b[ct] = *(const v16h*)(wrow + (ct * 16 + col) * ND + kb);
#pragma unroll
  for (int v = 0; v < 8; ++v) {
    int kk = (hi ? 16 : 0) + (v << 1);
    float2 xv = *(const float2*)(xrow + kb + kk);
    a[2 * v]     = (_Float16)xv.x;
    a[2 * v + 1] = (_Float16)xv.y;
  }
}

__device__ __forceinline__ void pa_mma(const v16h& a, const v16h b[4], v8f acc[4]) {
#pragma unroll
  for (int ct = 0; ct < 4; ++ct)
    acc[ct] = __builtin_amdgcn_wmma_f32_16x16x32_f16(
        false, a, false, b[ct], (short)0, acc[ct], false, false);
}

// ---------------------------------------------------------------------------
// Kernel 1 (pass A): hT[b][k][n] = sum_d x[b,n,d]*Wa[k,d], WMMA f16.
// One wave = 16 rows x 64 cols; D-loop in steps of 64 (2 chunks, ping-pong).
// ---------------------------------------------------------------------------
__global__ __launch_bounds__(256) void gat_pass_a(const float* __restrict__ x,
                                                  const _Float16* __restrict__ wa16,
                                                  _Float16* __restrict__ hT) {
  const int lane = threadIdx.x & 31;
  const int wave = threadIdx.x >> 5;
  const int tile = blockIdx.x * 8 + wave;          // 0..1023 row tiles of 16
  const int hi   = (lane & 16) ? 1 : 0;
  const int col  = lane & 15;
  const int row  = tile * 16 + col;                // global row in [0, B*N)
  const int b    = row >> 12;

  const float*    xrow = x + row * ND;
  const _Float16* wrow = wa16 + (hi ? 16 : 0);

  v8f acc[4];
#pragma unroll
  for (int ct = 0; ct < 4; ++ct) acc[ct] = (v8f){0.f,0.f,0.f,0.f,0.f,0.f,0.f,0.f};

  v16h aA, aB, bA[4], bB[4];
  pa_load(xrow, wrow, 0, hi, col, aA, bA);

  for (int kb = 0; kb < ND; kb += 64) {
    // stage 1: load kb+32 (always in range: <= ND-32), consume buffer A
    pa_load(xrow, wrow, kb + 32, hi, col, aB, bB);
    pa_mma(aA, bA, acc);
    // stage 2: load kb+64 (guarded), consume buffer B
    if (kb + 64 < ND) pa_load(xrow, wrow, kb + 64, hi, col, aA, bA);
    pa_mma(aB, bB, acc);
  }

  // D fragment: row = (hi?8:0)+v, col = lane&15  -> store transposed f16
#pragma unroll
  for (int ct = 0; ct < 4; ++ct) {
    const int kcol = ct * 16 + col;
#pragma unroll
    for (int v = 0; v < 8; ++v) {
      int rout = tile * 16 + (hi ? 8 : 0) + v;
      int nout = rout & (NN - 1);
      hT[((b * NK + kcol) << 12) + nout] = (_Float16)acc[ct][v];
    }
  }
}

// ---------------------------------------------------------------------------
// Kernel 2: t1/t2 per (b,n) from hT columns (coalesced across threads).
// ---------------------------------------------------------------------------
__global__ void gat_t1t2(const _Float16* __restrict__ hT,
                         const float* __restrict__ Wb, const float* __restrict__ wb_bias,
                         const float* __restrict__ Wc, const float* __restrict__ wc_bias,
                         float* __restrict__ t1, float* __restrict__ t2) {
  int gid = blockIdx.x * 256 + threadIdx.x;        // 0..16383
  int b = gid >> 12, n = gid & (NN - 1);
  float s1 = wb_bias[0], s2 = wc_bias[0];
  const _Float16* hp = hT + ((b * NK) << 12) + n;
#pragma unroll 8
  for (int k = 0; k < NK; ++k) {
    float hv = (float)hp[k << 12];
    s1 = fmaf(hv, Wb[k], s1);
    s2 = fmaf(hv, Wc[k], s2);
  }
  t1[gid] = s1;
  t2[gid] = s2;
}

// ---------------------------------------------------------------------------
// Kernel 3: per-batch max of t2 (for stable softmax).
// ---------------------------------------------------------------------------
__global__ void gat_t2max(const float* __restrict__ t2, float* __restrict__ t2mx) {
  __shared__ float red[256];
  int b = blockIdx.x;
  float m = -3.0e38f;
  for (int i = threadIdx.x; i < NN; i += 256) m = fmaxf(m, t2[b * NN + i]);
  red[threadIdx.x] = m;
  __syncthreads();
  for (int s = 128; s > 0; s >>= 1) {
    if (threadIdx.x < s) red[threadIdx.x] = fmaxf(red[threadIdx.x], red[threadIdx.x + s]);
    __syncthreads();
  }
  if (threadIdx.x == 0) t2mx[b] = red[0];
}

// ---------------------------------------------------------------------------
// Pass C helpers: B-fragment load from hT, weight-tile generation, WMMA set.
// ---------------------------------------------------------------------------
__device__ __forceinline__ void pc_load(const _Float16* __restrict__ hbase,
                                        int mb, v16h b[4]) {
#pragma unroll
  for (int ct = 0; ct < 4; ++ct)
    b[ct] = *(const v16h*)(hbase + ((ct * 16) << 12) + mb);
}

__device__ __forceinline__ void pc_prefetch(const _Float16* __restrict__ hbase, int mb) {
#pragma unroll
  for (int ct = 0; ct < 4; ++ct)
    __builtin_prefetch(hbase + ((ct * 16) << 12) + mb, 0, 0);
}

__device__ __forceinline__ void pc_weights(const float* __restrict__ t2s, int mb,
                                           int hi, float t1L, float rmax, v16h& a) {
#pragma unroll
  for (int e = 0; e < 16; ++e) {
    float s  = t1L + t2s[mb + a_frag_k(e, hi)];
    float ls = s > 0.f ? s : 0.2f * s;
    a[e] = (_Float16)__expf(ls - rmax);
  }
}

__device__ __forceinline__ void pc_mma(const v16h& a, const v16h b[4],
                                       const v16h& ones, v8f acc[4], v8f& accs) {
#pragma unroll
  for (int ct = 0; ct < 4; ++ct)
    acc[ct] = __builtin_amdgcn_wmma_f32_16x16x32_f16(
        false, a, false, b[ct], (short)0, acc[ct], false, false);
  accs = __builtin_amdgcn_wmma_f32_16x16x32_f16(
      false, a, false, ones, (short)0, accs, false, false);
}

// ---------------------------------------------------------------------------
// Kernel 4 (pass C): streaming weighted sum, ping-pong pipelined.
// One wave = 16 output rows x 64 cols; m-loop in steps of 64 (2 chunks).
// ---------------------------------------------------------------------------
__global__ __launch_bounds__(256) void gat_pass_c(const _Float16* __restrict__ hT,
                                                  const float* __restrict__ t1g,
                                                  const float* __restrict__ t2g,
                                                  const float* __restrict__ t2mxg,
                                                  const float* __restrict__ bias,
                                                  float* __restrict__ out) {
  __shared__ float t2s[NN];                        // 16 KB: whole t2 row for batch b
  const int lane = threadIdx.x & 31;
  const int wave = threadIdx.x >> 5;
  const int b    = blockIdx.x >> 5;                // 32 blocks per batch
  const int tile = (blockIdx.x & 31) * 8 + wave;   // 0..255 row tiles within batch
  const int hi   = (lane & 16) ? 1 : 0;
  const int col  = lane & 15;
  const int nrow = tile * 16 + col;

  for (int i = threadIdx.x; i < NN; i += 256) t2s[i] = t2g[b * NN + i];
  __syncthreads();

  const float t1L  = t1g[b * NN + nrow];
  const float smax = t1L + t2mxg[b];
  const float rmax = smax > 0.f ? smax : 0.2f * smax;   // lrelu of row max

  // per-lane base of B-fragment rows: hT[b][*][.], column set by lane
  const _Float16* hbase = hT + ((b * NK + col) << 12) + (hi ? 16 : 0);

  v8f acc[4];
#pragma unroll
  for (int ct = 0; ct < 4; ++ct) acc[ct] = (v8f){0.f,0.f,0.f,0.f,0.f,0.f,0.f,0.f};
  v8f accs = (v8f){0.f,0.f,0.f,0.f,0.f,0.f,0.f,0.f};

  v16h ones;
#pragma unroll
  for (int e = 0; e < 16; ++e) ones[e] = (_Float16)1.0f;

  v16h aA, aB, bA[4], bB[4];
  pc_load(hbase, 0, bA);
  pc_weights(t2s, 0, hi, t1L, rmax, aA);

  for (int mb = 0; mb < NN; mb += 64) {
    // stage 1: issue loads + weights for mb+32 (always in range), consume A
    pc_load(hbase, mb + 32, bB);
    pc_prefetch(hbase, mb + 96 < NN ? mb + 96 : mb + 32);
    pc_weights(t2s, mb + 32, hi, t1L, rmax, aB);
    pc_mma(aA, bA, ones, acc, accs);
    // stage 2: issue loads + weights for mb+64 (guarded), consume B
    if (mb + 64 < NN) {
      pc_load(hbase, mb + 64, bA);
      pc_prefetch(hbase, mb + 128 < NN ? mb + 128 : mb + 64);
      pc_weights(t2s, mb + 64, hi, t1L, rmax, aA);
    }
    pc_mma(aB, bB, ones, acc, accs);
  }

  // Epilogue: multiply by fast reciprocal of denominator, add bias, write f32.
#pragma unroll
  for (int ct = 0; ct < 4; ++ct) {
    const int kc = ct * 16 + col;
    const float bk = bias[kc];
#pragma unroll
    for (int v = 0; v < 8; ++v) {
      int nout = tile * 16 + (hi ? 8 : 0) + v;
      float inv = __builtin_amdgcn_rcpf(accs[v]);
      out[((b * NN + nout) << 6) + kc] = fmaf(acc[ct][v], inv, bk);
    }
  }
}

// ---------------------------------------------------------------------------
extern "C" void kernel_launch(void* const* d_in, const int* in_sizes, int n_in,
                              void* d_out, int out_size, void* d_ws, size_t ws_size,
                              hipStream_t stream) {
  const float* x       = (const float*)d_in[0];   // [B,N,D]
  const float* Wa      = (const float*)d_in[1];   // [K,D]
  const float* Wb      = (const float*)d_in[2];   // [K]
  const float* wb_bias = (const float*)d_in[3];   // scalar
  const float* Wc      = (const float*)d_in[4];   // [K]
  const float* wc_bias = (const float*)d_in[5];   // scalar
  const float* bias    = (const float*)d_in[6];   // [K]
  float* out = (float*)d_out;                      // [B,N,K] f32

  char* ws = (char*)d_ws;
  _Float16* hT   = (_Float16*)(ws + WS_HT);
  _Float16* wa16 = (_Float16*)(ws + WS_WA16);
  float*    t1   = (float*)(ws + WS_T1);
  float*    t2   = (float*)(ws + WS_T2);
  float*    t2mx = (float*)(ws + WS_T2MX);

  gat_cvt_wa<<<128, 256, 0, stream>>>(Wa, wa16);                  // 32768 elems
  gat_pass_a<<<128, 256, 0, stream>>>(x, wa16, hT);               // 1024 row tiles
  gat_t1t2  <<<64, 256, 0, stream>>>(hT, Wb, wb_bias, Wc, wc_bias, t1, t2);
  gat_t2max <<<NB, 256, 0, stream>>>(t2, t2mx);
  gat_pass_c<<<128, 256, 0, stream>>>(hT, t1, t2, t2mx, bias, out);
}
